// GATNet_7876970020894
// MI455X (gfx1250) — matrix operand order repaired and checked by reading, not compiled, not run on hardware
//
#include <hip/hip_runtime.h>
#include <math.h>

typedef __attribute__((ext_vector_type(16))) __bf16 v16bf;
typedef __attribute__((ext_vector_type(8)))  float  v8f;

#define GN 100000
#define GE 1600000
#define SLOPE 0.2f

// ---------------- utility kernels ----------------

__global__ void fill_kernel(float* __restrict__ p, float v, int n) {
    int t = blockIdx.x * blockDim.x + threadIdx.x;
    if (t < n) p[t] = v;
}

// float atomic max via signed-int max / unsigned-int min trick (valid with -inf init)
__device__ __forceinline__ void atomicMaxF32(float* addr, float val) {
    if (val >= 0.0f) atomicMax((int*)addr, __float_as_int(val));
    else             atomicMin((unsigned int*)addr, __float_as_uint(val));
}

// ---------------- WMMA GEMM: Out[nrows,NCOL] = A[nrows,K] * B[K,NCOL] (bf16 in, f32 acc) ----------------

template<int K, int NCOL>
__global__ void __launch_bounds__(256)
gemm_wmma_kernel(const float* __restrict__ A, const float* __restrict__ B,
                 float* __restrict__ Out, int nrows) {
    constexpr int KCH = K / 32;    // k-chunks of 32
    constexpr int NT  = NCOL / 16; // 16-wide column tiles

    __shared__ __align__(32) __bf16 bfrag[KCH][NT][32][16];

    const int tid = threadIdx.x;

    // Stage B fragments into LDS in exact WMMA B layout:
    // lane l: col = (l&15), k = (l<16 ? 0 : 8) + (e&7) + (e<8 ? 0 : 16)  (e = half index 0..15)
    for (int idx = tid; idx < KCH * NT * 32 * 16; idx += 256) {
        int e    = idx & 15;
        int lane = (idx >> 4) & 31;
        int nt   = (idx >> 9) % NT;
        int kc   = idx / (512 * NT);
        int k    = ((lane < 16) ? 0 : 8) + (e & 7) + ((e < 8) ? 0 : 16);
        int col  = nt * 16 + (lane & 15);
        bfrag[kc][nt][lane][e] = (__bf16)B[(kc * 32 + k) * NCOL + col];
    }
    __syncthreads();

    const int wave = tid >> 5;
    const int lane = tid & 31;
    const int row0 = blockIdx.x * 128 + wave * 16;
    int row = row0 + (lane & 15);
    if (row > nrows - 1) row = nrows - 1;      // clamp reads; writes guarded below
    const int kbase = (lane < 16) ? 0 : 8;     // A-matrix half-wave K split

    v8f acc[NT] = {};

    #pragma unroll
    for (int kc = 0; kc < KCH; kc++) {
        // A 16x32 bf16 layout: lane holds K in runs [kbase..kbase+7] and [kbase+16..kbase+23]
        const float* ap = A + (size_t)row * K + kc * 32 + kbase;
        v16bf a;
        #pragma unroll
        for (int e = 0; e < 8; e++) {
            a[e]     = (__bf16)ap[e];
            a[e + 8] = (__bf16)ap[e + 16];
        }
        #pragma unroll
        for (int nt = 0; nt < NT; nt++) {
            v16bf b = *reinterpret_cast<const v16bf*>(&bfrag[kc][nt][lane][0]);
            acc[nt] = __builtin_amdgcn_wmma_f32_16x16x32_bf16(
                false, a, false, b, (short)0, acc[nt], false, false);
        }
    }

    // C/D layout: col = lane&15 ; row = row0 + v + (lane>=16 ? 8 : 0)
    const int rbase = row0 + ((lane >= 16) ? 8 : 0);
    #pragma unroll
    for (int nt = 0; nt < NT; nt++) {
        int col = nt * 16 + (lane & 15);
        #pragma unroll
        for (int v = 0; v < 8; v++) {
            int r = rbase + v;
            if (r < nrows) Out[(size_t)r * NCOL + col] = acc[nt][v];
        }
    }
}

// ---------------- attention score reduction: s[n,h] = sum_c h[n,h,c]*a_src[h,c] ----------------

__global__ void attn_kernel(const float* __restrict__ h, const float* __restrict__ asrc,
                            const float* __restrict__ adst, float* __restrict__ s,
                            float* __restrict__ d, int n, int H, int C) {
    int t = blockIdx.x * blockDim.x + threadIdx.x;
    if (t >= n * H) return;
    int hd = t % H;
    int node = t / H;
    const float* hp = h + (size_t)node * H * C + hd * C;
    float ss = 0.f, dd = 0.f;
    for (int c = 0; c < C; c++) {
        float v = hp[c];
        ss += v * asrc[hd * C + c];
        dd += v * adst[hd * C + c];
    }
    s[t] = ss;
    d[t] = dd;
}

// ---------------- edge passes (edges 0..E-1 from edge_index, E..E+N-1 are self loops) ----------------

__device__ __forceinline__ void edge_ends(const int* ei, int eidx, int& src, int& dst) {
    if (eidx < GE) { src = ei[eidx]; dst = ei[GE + eidx]; }
    else           { src = eidx - GE; dst = eidx - GE; }
}

__global__ void edge_max_kernel(const int* __restrict__ ei, const float* __restrict__ s,
                                const float* __restrict__ d, float* __restrict__ m, int H) {
    int t = blockIdx.x * blockDim.x + threadIdx.x;
    int total = (GE + GN) * H;
    if (t >= total) return;
    int hd = t % H, eidx = t / H;
    int src, dst; edge_ends(ei, eidx, src, dst);
    float e = s[src * H + hd] + d[dst * H + hd];
    e = (e > 0.f) ? e : SLOPE * e;
    atomicMaxF32(&m[dst * H + hd], e);
}

__global__ void edge_sum_kernel(const int* __restrict__ ei, const float* __restrict__ s,
                                const float* __restrict__ d, const float* __restrict__ m,
                                float* __restrict__ z, int H) {
    int t = blockIdx.x * blockDim.x + threadIdx.x;
    int total = (GE + GN) * H;
    if (t >= total) return;
    int hd = t % H, eidx = t / H;
    int src, dst; edge_ends(ei, eidx, src, dst);
    float e = s[src * H + hd] + d[dst * H + hd];
    e = (e > 0.f) ? e : SLOPE * e;
    float ex = __expf(e - m[dst * H + hd]);
    atomicAdd(&z[dst * H + hd], ex);
}

__global__ void edge_agg_kernel(const int* __restrict__ ei, const float* __restrict__ s,
                                const float* __restrict__ d, const float* __restrict__ m,
                                const float* __restrict__ z, const float* __restrict__ h,
                                float* __restrict__ agg, int H, int C, int CT) {
    int t = blockIdx.x * blockDim.x + threadIdx.x;
    int ngrp = C / CT;
    int total = (GE + GN) * H * ngrp;
    if (t >= total) return;
    int g = t % ngrp;
    int t2 = t / ngrp;
    int hd = t2 % H, eidx = t2 / H;
    int src, dst; edge_ends(ei, eidx, src, dst);
    float e = s[src * H + hd] + d[dst * H + hd];
    e = (e > 0.f) ? e : SLOPE * e;
    float ex = __expf(e - m[dst * H + hd]);
    float alpha = ex / (z[dst * H + hd] + 1e-16f);
    const float* hp = h   + (size_t)src * H * C + hd * C + g * CT;
    float*       ap = agg + (size_t)dst * H * C + hd * C + g * CT;
    for (int c = 0; c < CT; c++) atomicAdd(&ap[c], alpha * hp[c]);
}

// ---------------- epilogues ----------------

__global__ void bias_elu_kernel(float* __restrict__ x, const float* __restrict__ b, int n, int F) {
    int t = blockIdx.x * blockDim.x + threadIdx.x;
    if (t >= n * F) return;
    float v = x[t] + b[t % F];
    x[t] = (v > 0.f) ? v : (__expf(v) - 1.0f);
}

__global__ void logsoftmax_kernel(const float* __restrict__ agg, const float* __restrict__ b,
                                  float* __restrict__ out, int n) {
    int t = blockIdx.x * blockDim.x + threadIdx.x;
    if (t >= n) return;
    float v[32];
    float mx = -INFINITY;
    #pragma unroll
    for (int c = 0; c < 32; c++) {
        v[c] = agg[(size_t)t * 32 + c] + b[c];
        mx = fmaxf(mx, v[c]);
    }
    float sum = 0.f;
    #pragma unroll
    for (int c = 0; c < 32; c++) sum += __expf(v[c] - mx);
    float ls = __logf(sum) + mx;
    #pragma unroll
    for (int c = 0; c < 32; c++) out[(size_t)t * 32 + c] = v[c] - ls;
}

// ---------------- host launch ----------------

static inline int cdiv(long long a, int b) { return (int)((a + b - 1) / b); }

extern "C" void kernel_launch(void* const* d_in, const int* in_sizes, int n_in,
                              void* d_out, int out_size, void* d_ws, size_t ws_size,
                              hipStream_t stream) {
    const float* x   = (const float*)d_in[0];
    const int*   ei  = (const int*)  d_in[1];
    const float* W1  = (const float*)d_in[2];
    const float* as1 = (const float*)d_in[3];
    const float* ad1 = (const float*)d_in[4];
    const float* b1  = (const float*)d_in[5];
    const float* W2  = (const float*)d_in[6];
    const float* as2 = (const float*)d_in[7];
    const float* ad2 = (const float*)d_in[8];
    const float* b2  = (const float*)d_in[9];
    float* out = (float*)d_out;

    float* ws  = (float*)d_ws;
    float* h1  = ws;                      // N*64
    float* s   = h1 + (size_t)GN * 64;    // N*8
    float* dd  = s  + (size_t)GN * 8;     // N*8
    float* m   = dd + (size_t)GN * 8;     // N*8
    float* z   = m  + (size_t)GN * 8;     // N*8
    float* agg = z  + (size_t)GN * 8;     // N*64
    // layer-2 reuse
    float* h2   = h1;                     // N*32 (disjoint from agg)
    float* agg2 = h1 + (size_t)GN * 32;   // N*32

    const int B = 256;
    const int TOT = GE + GN;

    // ===== layer 1 =====
    fill_kernel<<<cdiv(GN * 8, B), B, 0, stream>>>(m, -INFINITY, GN * 8);
    fill_kernel<<<cdiv(GN * 8, B), B, 0, stream>>>(z, 0.0f, GN * 8);
    fill_kernel<<<cdiv((long long)GN * 64, B), B, 0, stream>>>(agg, 0.0f, GN * 64);

    gemm_wmma_kernel<256, 64><<<cdiv(GN, 128), B, 0, stream>>>(x, W1, h1, GN);
    attn_kernel<<<cdiv(GN * 8, B), B, 0, stream>>>(h1, as1, ad1, s, dd, GN, 8, 8);

    edge_max_kernel<<<cdiv((long long)TOT * 8, B), B, 0, stream>>>(ei, s, dd, m, 8);
    edge_sum_kernel<<<cdiv((long long)TOT * 8, B), B, 0, stream>>>(ei, s, dd, m, z, 8);
    edge_agg_kernel<<<cdiv((long long)TOT * 8, B), B, 0, stream>>>(ei, s, dd, m, z, h1, agg, 8, 8, 8);

    bias_elu_kernel<<<cdiv((long long)GN * 64, B), B, 0, stream>>>(agg, b1, GN, 64);

    // ===== layer 2 =====
    fill_kernel<<<cdiv(GN, B), B, 0, stream>>>(m, -INFINITY, GN);
    fill_kernel<<<cdiv(GN, B), B, 0, stream>>>(z, 0.0f, GN);
    fill_kernel<<<cdiv((long long)GN * 32, B), B, 0, stream>>>(agg2, 0.0f, GN * 32);

    gemm_wmma_kernel<64, 32><<<cdiv(GN, 128), B, 0, stream>>>(agg, W2, h2, GN);
    attn_kernel<<<cdiv(GN, B), B, 0, stream>>>(h2, as2, ad2, s, dd, GN, 1, 32);

    edge_max_kernel<<<cdiv((long long)TOT, B), B, 0, stream>>>(ei, s, dd, m, 1);
    edge_sum_kernel<<<cdiv((long long)TOT, B), B, 0, stream>>>(ei, s, dd, m, z, 1);
    edge_agg_kernel<<<cdiv((long long)TOT * 4, B), B, 0, stream>>>(ei, s, dd, m, z, h2, agg2, 1, 32, 8);

    logsoftmax_kernel<<<cdiv(GN, B), B, 0, stream>>>(agg2, b2, out, GN);
}
